// SelfAttention_69776038691343
// MI455X (gfx1250) — matrix operand order repaired and checked by reading, not compiled, not run on hardware
//
#include <hip/hip_runtime.h>
#include <math.h>

// ---------------------------------------------------------------------------
// Self-attention forward for MI455X (gfx1250), bf16 WMMA everywhere.
// B=2, N=2048, DIM=1024, HEADS=16, HEAD_DIM=64.
// ---------------------------------------------------------------------------

typedef __attribute__((ext_vector_type(16))) __bf16 v16bf;
typedef __attribute__((ext_vector_type(8)))  __bf16 bf16x8;
typedef __attribute__((ext_vector_type(4)))  __bf16 bf16x4;
typedef __attribute__((ext_vector_type(8)))  float  v8f;

#define DIMM   1024
#define HEADS  16
#define HD     64
#define BATCH  2
#define SEQ    2048
#define ROWS   (BATCH * SEQ)      // 4096

// ---- async global->LDS (CDNA5 GLOBAL_LOAD_ASYNC_TO_LDS_B128) --------------
#if defined(__has_builtin)
#if __has_builtin(__builtin_amdgcn_global_load_async_to_lds_b128)
#define HAVE_ASYNC_LDS 1
#endif
#endif
#ifndef HAVE_ASYNC_LDS
#define HAVE_ASYNC_LDS 0
#endif

typedef int v4i __attribute__((vector_size(16)));
typedef __attribute__((address_space(1))) v4i* gptr_v4i;   // global int4*
typedef __attribute__((address_space(3))) v4i* lptr_v4i;   // LDS int4*

__device__ __forceinline__ void copy16_g2l(const void* g, void* l) {
#if HAVE_ASYNC_LDS
  __builtin_amdgcn_global_load_async_to_lds_b128((gptr_v4i)g, (lptr_v4i)l, 0, 0);
#else
  *(uint4*)l = *(const uint4*)g;
#endif
}
__device__ __forceinline__ void async_join() {
#if HAVE_ASYNC_LDS
  asm volatile("s_wait_asynccnt 0" ::: "memory");
#endif
}

// ---- WMMA fragment loaders from LDS (row-major staging) -------------------
// A operand (16x32 bf16): lane half h holds two contiguous 8-elem chunks:
//   k in [kbase + h*8, +8) and [kbase + 16 + h*8, +8)   (per ISA 7.12.2)
__device__ __forceinline__ v16bf ldsA_frag(const __bf16* row_ptr, int kbase, int half) {
  union { v16bf v; bf16x8 h[2]; } u;
  u.h[0] = *(const bf16x8*)(row_ptr + kbase + half * 8);
  u.h[1] = *(const bf16x8*)(row_ptr + kbase + 16 + half * 8);
  return u.v;
}
// B operand (32x16 bf16), staged as [n][k] row-major: lane half h holds
// k in [kbase + h*8, +8) and [kbase + 16 + h*8, +8) for its column n = lane%16.
__device__ __forceinline__ v16bf ldsB_frag(const __bf16* col_ptr, int kbase, int half) {
  union { v16bf v; bf16x8 h[2]; } u;
  u.h[0] = *(const bf16x8*)(col_ptr + kbase + half * 8);
  u.h[1] = *(const bf16x8*)(col_ptr + kbase + 16 + half * 8);
  return u.v;
}

// ---------------------------------------------------------------------------
// fp32 -> bf16 conversion (bandwidth-bound, vectorized x4)
// ---------------------------------------------------------------------------
__global__ __launch_bounds__(256)
void cvt_f32_to_bf16(const float* __restrict__ in, __bf16* __restrict__ out, int n) {
  int i = (blockIdx.x * 256 + threadIdx.x) * 4;
  if (i + 4 <= n) {
    float4 v = *(const float4*)(in + i);
    bf16x4 o;
    o.x = (__bf16)v.x; o.y = (__bf16)v.y; o.z = (__bf16)v.z; o.w = (__bf16)v.w;
    *(bf16x4*)(out + i) = o;
  }
}

// ---------------------------------------------------------------------------
// Tiled bf16 WMMA GEMM: C[M,N] = A[M,K] @ B[K,N] + bias
// Block: 256 threads (8 waves). Block tile 256(M) x 64(N), K-step 64.
// Wave w computes rows [w*32, w*32+32) x all 64 cols -> 8 accum tiles,
// 16 wmma per stage round (4 A-frags + 8 B-frags reloaded from LDS).
// ---------------------------------------------------------------------------
__global__ __launch_bounds__(256)
void gemm_bf16_wmma(const __bf16* __restrict__ A, const __bf16* __restrict__ B,
                    const float* __restrict__ bias, void* __restrict__ C,
                    int M, int N, int K, int c_is_f32) {
  __shared__ alignas(32) __bf16 Asub[256 * 64];   // [row][k]  32 KB
  __shared__ alignas(32) __bf16 Bt[64 * 64];      // [n][k]     8 KB

  const int tid  = threadIdx.x;
  const int w    = tid >> 5;
  const int lane = tid & 31;
  const int half = lane >> 4;
  const int lr   = lane & 15;
  const int mb   = blockIdx.y * 256;
  const int nb   = blockIdx.x * 64;

  v8f acc[2][4] = {};

  for (int kk = 0; kk < K; kk += 64) {
    if (kk + 64 < K) {  // speculative prefetch of next B tile (global_prefetch_b8)
      __builtin_prefetch(B + (size_t)(kk + 64 + (tid >> 2)) * N + nb + (tid & 3) * 16, 0, 1);
    }
    // stage A tile 256x64 row-major: 2048 b128 transfers, async to LDS
#pragma unroll
    for (int i = 0; i < 8; ++i) {
      int t = tid + i * 256;
      int r = t >> 3, seg = (t & 7) * 8;
      copy16_g2l(A + (size_t)(mb + r) * K + kk + seg, &Asub[r * 64 + seg]);
    }
    // stage B tile 64x64 transposed -> Bt[n][k] (coalesced global reads)
#pragma unroll
    for (int i = 0; i < 4; ++i) {
      int t = tid + i * 256;
      int k = t >> 4, ng = (t & 15) * 4;
      bf16x4 v = *(const bf16x4*)(B + (size_t)(kk + k) * N + nb + ng);
      Bt[(ng + 0) * 64 + k] = v.x;
      Bt[(ng + 1) * 64 + k] = v.y;
      Bt[(ng + 2) * 64 + k] = v.z;
      Bt[(ng + 3) * 64 + k] = v.w;
    }
    async_join();
    __syncthreads();

#pragma unroll
    for (int kb = 0; kb < 64; kb += 32) {
      v16bf af0 = ldsA_frag(&Asub[(w * 32 + lr) * 64], kb, half);
      v16bf af1 = ldsA_frag(&Asub[(w * 32 + 16 + lr) * 64], kb, half);
#pragma unroll
      for (int nt = 0; nt < 4; ++nt) {
        v16bf bf = ldsB_frag(&Bt[(nt * 16 + lr) * 64], kb, half);
        acc[0][nt] = __builtin_amdgcn_wmma_f32_16x16x32_bf16(
            false, af0, false, bf, (short)0, acc[0][nt], false, false);
        acc[1][nt] = __builtin_amdgcn_wmma_f32_16x16x32_bf16(
            false, af1, false, bf, (short)0, acc[1][nt], false, false);
      }
    }
    __syncthreads();
  }

  // epilogue: bias + store (D layout: vgpr r -> row r + 8*half, col = lr)
#pragma unroll
  for (int mi = 0; mi < 2; ++mi)
#pragma unroll
    for (int nt = 0; nt < 4; ++nt) {
      int col = nb + nt * 16 + lr;
      float bv = bias ? bias[col] : 0.f;
#pragma unroll
      for (int r = 0; r < 8; ++r) {
        int row = mb + w * 32 + mi * 16 + r + 8 * half;
        float v = acc[mi][nt][r] + bv;
        if (c_is_f32) ((float*)C)[(size_t)row * N + col] = v;
        else          ((__bf16*)C)[(size_t)row * N + col] = (__bf16)v;
      }
    }
}

// ---------------------------------------------------------------------------
// Flash-attention forward. One block = one (b,h) and 128 query rows.
// 8 waves; wave w owns query rows [w*16, w*16+16). Key tiles of 64.
// QKV is bf16 row-major [4096, 3072] (q|k|v per row). CTX bf16 [4096,1024].
// ---------------------------------------------------------------------------
__global__ __launch_bounds__(256)
void attn_fwd(const __bf16* __restrict__ QKV, __bf16* __restrict__ CTX) {
  __shared__ alignas(32) __bf16 Qs[128 * 64];     // [qrow][d]  16 KB
  __shared__ alignas(32) __bf16 Ks[64 * 64];      // [key][d]    8 KB
  __shared__ alignas(32) __bf16 Vt[64 * 64];      // [d][key]    8 KB
  __shared__ alignas(32) __bf16 Ps[8 * 16 * 64];  // per-wave P 16 KB

  const int tid  = threadIdx.x;
  const int w    = tid >> 5;
  const int lane = tid & 31;
  const int half = lane >> 4;
  const int lr   = lane & 15;
  const int bh   = blockIdx.y;
  const int b    = bh >> 4;
  const int h    = bh & 15;
  const int q0   = blockIdx.x * 128;
  const int ldq  = 3 * DIMM;  // 3072

  const __bf16* Qg = QKV + (size_t)b * SEQ * ldq + (size_t)h * HD;
  const __bf16* Kg = Qg + DIMM;
  const __bf16* Vg = Qg + 2 * DIMM;

  // stage Q tile 128x64 once (async; joined before first compute)
#pragma unroll
  for (int i = 0; i < 4; ++i) {
    int t = tid + i * 256;
    int r = t >> 3, seg = (t & 7) * 8;
    copy16_g2l(Qg + (size_t)(q0 + r) * ldq + seg, &Qs[r * 64 + seg]);
  }

  v8f o[4] = {};
  float mrow[8], lrow[8];
#pragma unroll
  for (int r = 0; r < 8; ++r) { mrow[r] = -1e30f; lrow[r] = 0.f; }

  __bf16* myP = &Ps[w * 16 * 64];
  const float scale = 0.125f;  // 1/sqrt(64)

  for (int kt = 0; kt < SEQ / 64; ++kt) {
    __syncthreads();  // prior iter done reading Ks/Vt
    const int k0 = kt * 64;
    // stage K tile row-major [key][d] (async)
#pragma unroll
    for (int i = 0; i < 2; ++i) {
      int t = tid + i * 256;
      int r = t >> 3, seg = (t & 7) * 8;
      copy16_g2l(Kg + (size_t)(k0 + r) * ldq + seg, &Ks[r * 64 + seg]);
    }
    // stage V tile transposed -> Vt[d][key]
#pragma unroll
    for (int i = 0; i < 2; ++i) {
      int t = tid + i * 256;
      int key = t >> 3, seg = (t & 7) * 8;
      bf16x8 v = *(const bf16x8*)(Vg + (size_t)(k0 + key) * ldq + seg);
#pragma unroll
      for (int j = 0; j < 8; ++j) Vt[(seg + j) * 64 + key] = v[j];
    }
    async_join();
    __syncthreads();

    // S = Q @ K^T  (head-dim K=64 -> two wmma per 16x16 tile)
    v8f s[4] = {};
    v16bf aq0 = ldsA_frag(&Qs[(w * 16 + lr) * 64], 0, half);
    v16bf aq1 = ldsA_frag(&Qs[(w * 16 + lr) * 64], 32, half);
#pragma unroll
    for (int nt = 0; nt < 4; ++nt) {
      v16bf bk0 = ldsB_frag(&Ks[(nt * 16 + lr) * 64], 0, half);
      v16bf bk1 = ldsB_frag(&Ks[(nt * 16 + lr) * 64], 32, half);
      s[nt] = __builtin_amdgcn_wmma_f32_16x16x32_bf16(
          false, aq0, false, bk0, (short)0, s[nt], false, false);
      s[nt] = __builtin_amdgcn_wmma_f32_16x16x32_bf16(
          false, aq1, false, bk1, (short)0, s[nt], false, false);
    }
#pragma unroll
    for (int nt = 0; nt < 4; ++nt)
#pragma unroll
      for (int r = 0; r < 8; ++r) s[nt][r] *= scale;

    // online softmax. Row (r + 8*half) lives across the 16 lanes of this
    // half-wave -> shfl_xor masks 1..8 reduce within the group.
    float alpha[8];
#pragma unroll
    for (int r = 0; r < 8; ++r) {
      float mx = fmaxf(fmaxf(s[0][r], s[1][r]), fmaxf(s[2][r], s[3][r]));
#pragma unroll
      for (int off = 1; off < 16; off <<= 1)
        mx = fmaxf(mx, __shfl_xor(mx, off, 32));
      float mnew = fmaxf(mrow[r], mx);
      alpha[r] = __expf(mrow[r] - mnew);
      mrow[r] = mnew;
    }
#pragma unroll
    for (int r = 0; r < 8; ++r) {
      float rs = 0.f;
#pragma unroll
      for (int nt = 0; nt < 4; ++nt) {
        float p = __expf(s[nt][r] - mrow[r]);
        s[nt][r] = p;
        rs += p;
      }
#pragma unroll
      for (int off = 1; off < 16; off <<= 1)
        rs += __shfl_xor(rs, off, 32);
      lrow[r] = lrow[r] * alpha[r] + rs;
#pragma unroll
      for (int dt = 0; dt < 4; ++dt) o[dt][r] *= alpha[r];
    }

    // stage P (bf16) to this wave's private LDS region, D-layout -> row-major
#pragma unroll
    for (int nt = 0; nt < 4; ++nt)
#pragma unroll
      for (int r = 0; r < 8; ++r)
        myP[(r + 8 * half) * 64 + nt * 16 + lr] = (__bf16)s[nt][r];
    asm volatile("s_wait_dscnt 0" ::: "memory");  // per-wave LDS RAW fence

    // O += P @ V   (A = P 16x64, B = V via Vt[d][key])
    v16bf ap0 = ldsA_frag(&myP[lr * 64], 0, half);
    v16bf ap1 = ldsA_frag(&myP[lr * 64], 32, half);
#pragma unroll
    for (int dt = 0; dt < 4; ++dt) {
      v16bf bv0 = ldsB_frag(&Vt[(dt * 16 + lr) * 64], 0, half);
      v16bf bv1 = ldsB_frag(&Vt[(dt * 16 + lr) * 64], 32, half);
      o[dt] = __builtin_amdgcn_wmma_f32_16x16x32_bf16(
          false, ap0, false, bv0, (short)0, o[dt], false, false);
      o[dt] = __builtin_amdgcn_wmma_f32_16x16x32_bf16(
          false, ap1, false, bv1, (short)0, o[dt], false, false);
    }
  }

  // finalize: O /= l, store ctx bf16 [B*SEQ, DIM]
#pragma unroll
  for (int r = 0; r < 8; ++r) {
    float inv = 1.f / lrow[r];
    int row = q0 + w * 16 + r + 8 * half;
    size_t base = ((size_t)b * SEQ + row) * DIMM + (size_t)h * HD;
#pragma unroll
    for (int dt = 0; dt < 4; ++dt)
      CTX[base + dt * 16 + lr] = (__bf16)(o[dt][r] * inv);
  }
}

// ---------------------------------------------------------------------------
extern "C" void kernel_launch(void* const* d_in, const int* in_sizes, int n_in,
                              void* d_out, int out_size, void* d_ws, size_t ws_size,
                              hipStream_t stream) {
  const float* x     = (const float*)d_in[0];
  const float* W_qkv = (const float*)d_in[1];
  const float* b_qkv = (const float*)d_in[2];
  const float* W_out = (const float*)d_in[3];
  const float* b_out = (const float*)d_in[4];

  // workspace layout (bf16 buffers), ~50.3 MB total
  char* ws = (char*)d_ws;
  __bf16* Xbf    = (__bf16*)ws; ws += (size_t)ROWS * DIMM * 2;        // 8 MB
  __bf16* Wqkvbf = (__bf16*)ws; ws += (size_t)DIMM * 3 * DIMM * 2;    // 6 MB
  __bf16* Woutbf = (__bf16*)ws; ws += (size_t)DIMM * DIMM * 2;        // 2 MB
  __bf16* QKVbf  = (__bf16*)ws; ws += (size_t)ROWS * 3 * DIMM * 2;    // 24 MB
  __bf16* CTXbf  = (__bf16*)ws;                                       // 8 MB

  // 1) fp32 -> bf16 conversions
  {
    int n = ROWS * DIMM;
    cvt_f32_to_bf16<<<(n / 4 + 255) / 256, 256, 0, stream>>>(x, Xbf, n);
    n = DIMM * 3 * DIMM;
    cvt_f32_to_bf16<<<(n / 4 + 255) / 256, 256, 0, stream>>>(W_qkv, Wqkvbf, n);
    n = DIMM * DIMM;
    cvt_f32_to_bf16<<<(n / 4 + 255) / 256, 256, 0, stream>>>(W_out, Woutbf, n);
  }

  // 2) QKV projection: [4096,3072] bf16 = Xbf @ Wqkv + b_qkv
  {
    dim3 grid((3 * DIMM) / 64, ROWS / 256);  // (48, 16)
    gemm_bf16_wmma<<<grid, 256, 0, stream>>>(Xbf, Wqkvbf, b_qkv, QKVbf,
                                             ROWS, 3 * DIMM, DIMM, 0);
  }

  // 3) flash attention -> CTX bf16 [4096, 1024]
  {
    dim3 grid(SEQ / 128, BATCH * HEADS);  // (16, 32)
    attn_fwd<<<grid, 256, 0, stream>>>(QKVbf, CTXbf);
  }

  // 4) output projection: d_out fp32 = CTX @ Wout + b_out
  {
    dim3 grid(DIMM / 64, ROWS / 256);  // (16, 16)
    gemm_bf16_wmma<<<grid, 256, 0, stream>>>(CTXbf, Woutbf, b_out, d_out,
                                             ROWS, DIMM, DIMM, 1);
  }
}
